// LongSeqClassifier_73461120631139
// MI455X (gfx1250) — matrix operand order repaired and checked by reading, not compiled, not run on hardware
//
#include <hip/hip_runtime.h>
#include <hip/hip_bf16.h>
#include <stdint.h>

// ---------------------------------------------------------------------------
// CDNA5 (gfx1250) hierarchical BiLSTM classifier.
// bf16 V_WMMA_F32_16X16X32_BF16 for all matrix work, f32 accumulation.
// xg is produced/consumed in C-fragment-native blocked layout
// [mtile][ntile][lane][8] with t-major logical rows, so the sequential scan
// loads its accumulators with one 32B v8f per tile per lane.
// The scan re-streams W_hh from L2 every step; an opaque offset defeats the
// compiler's invariant-load hoisting (which otherwise spills 512 VGPRs of
// weight fragments to scratch).
// ---------------------------------------------------------------------------

typedef __attribute__((ext_vector_type(16))) __bf16         v16bf;
typedef __attribute__((ext_vector_type(16))) unsigned short v16u;
typedef __attribute__((ext_vector_type(8)))  unsigned short v8u;
typedef __attribute__((ext_vector_type(8)))  float          v8f;
typedef __attribute__((ext_vector_type(4)))  float          v4f;

__device__ __forceinline__ unsigned short f32_to_bf16_rne(float f) {
  unsigned int u = __float_as_uint(f);
  u += 0x7FFFu + ((u >> 16) & 1u);          // round-to-nearest-even
  return (unsigned short)(u >> 16);
}

__device__ __forceinline__ float sigmoidf_fast(float x) {
  return 1.0f / (1.0f + __expf(-x));
}

__device__ __forceinline__ v8f wmma_bf16_f32(v16u a, v16u b, v8f c) {
  return __builtin_amdgcn_wmma_f32_16x16x32_bf16(
      false, __builtin_bit_cast(v16bf, a),
      false, __builtin_bit_cast(v16bf, b),
      (short)0, c, false, false);
}

// A fragment (16x32) split-pattern load: elements 0..7 at p, 8..15 at p+16
// (p already includes the lane-dependent +0/+8 half-select).
__device__ __forceinline__ v16u load_frag_pair(const unsigned short* p) {
  v8u lo = *(const v8u*)(p);
  v8u hi = *(const v8u*)(p + 16);
  v16u out;
#pragma unroll
  for (int i = 0; i < 8; ++i) { out[i] = lo[i]; out[8 + i] = hi[i]; }
  return out;
}

// ---------------------------------------------------------------------------
// Kernel: vectorized f32 -> bf16 (8 elements / thread); n must be %8 == 0.
// ---------------------------------------------------------------------------
__global__ void k_f32_to_bf16x8(const float* __restrict__ src,
                                unsigned short* __restrict__ dst, int n8) {
  int i = blockIdx.x * blockDim.x + threadIdx.x;
  if (i >= n8) return;
  const float* p = src + (long)i * 8;
  v4f a = *(const v4f*)(p);
  v4f b = *(const v4f*)(p + 4);
  v8u o;
#pragma unroll
  for (int j = 0; j < 4; ++j) {
    o[j]     = f32_to_bf16_rne(a[j]);
    o[4 + j] = f32_to_bf16_rne(b[j]);
  }
  *(v8u*)(dst + (long)i * 8) = o;
}

// ---------------------------------------------------------------------------
// Kernel: blocked-output GEMM, templated on K (compile-time tile offsets).
//   logical rows are t-major: mlog = t*nSeq + seq ; A row in Xbf storage is
//   min(seq, validSeq-1)*T + t  (clamp pads sentence level to 16 seqs).
//   OUT[mlog][n] = Xbf[row] @ W[n][:] + bias[n], written as
//   outBlk[mtile][n/16][lane][8]  (C-fragment native, v8f per tile per lane).
// block = 256 threads (8 waves); wave covers 8 N-tiles => block = 16 x 1024.
// grid.x = nSeq*T/16.   N fixed at 1024.
// ---------------------------------------------------------------------------
template <int K>
__global__ void k_gemm_bf16_blk(const unsigned short* __restrict__ Xbf,
                                const unsigned short* __restrict__ W,
                                const float* __restrict__ bias,
                                float* __restrict__ outBlk,
                                int nSeq, int T, int validSeq) {
  const int lane  = threadIdx.x & 31;
  const int wave  = threadIdx.x >> 5;
  const int mtile = blockIdx.x;
  const int nt0   = wave * 8;

  int mlog = mtile * 16 + (lane & 15);
  int seq  = mlog % nSeq;
  int t    = mlog / nSeq;
  if (seq >= validSeq) seq = validSeq - 1;
  const unsigned short* aptr = Xbf + (long)(seq * T + t) * K + ((lane >> 4) << 3);
  // Single B base pointer; per-tile offsets i*16*K are compile-time constants.
  const unsigned short* wptr = W + (long)(nt0 * 16 + (lane & 15)) * K
                                 + ((lane >> 4) << 3);

  v8f acc[8];
#pragma unroll
  for (int i = 0; i < 8; ++i) acc[i] = v8f{};

  v16u a = load_frag_pair(aptr);
  for (int kb = 0; kb < K; kb += 32) {
    // Issue all B loads for this k-step, then next A, then the WMMAs.
    v16u b[8];
#pragma unroll
    for (int i = 0; i < 8; ++i) b[i] = load_frag_pair(wptr + i * 16 * K);
    v16u a_next;
    if (kb + 32 < K) a_next = load_frag_pair(aptr + 32);
#pragma unroll
    for (int i = 0; i < 8; ++i) acc[i] = wmma_bf16_f32(a, b[i], acc[i]);
    a = a_next;
    aptr += 32;
    wptr += 32;
  }

#pragma unroll
  for (int i = 0; i < 8; ++i) {
    float bv = bias[(nt0 + i) * 16 + (lane & 15)];
#pragma unroll
    for (int r = 0; r < 8; ++r) acc[i][r] += bv;
    long off = (((long)mtile * 64 + nt0 + i) * 32 + lane) << 3;
    *(v8f*)(outBlk + off) = acc[i];
  }
}

// ---------------------------------------------------------------------------
// Kernel: LSTM scan, 16 sequences per block, one direction.
//   xgBlk : blocked preactivations, mtile(t) = t*nSeqTiles + blockIdx.x
//   whh   : [1024][256] bf16 row-major
// Wave w owns hidden slice j in [32w,32w+32) across all four gates -> gate
// math is element-wise in C-fragment registers; only h goes through LDS.
// W_hh fragments are deliberately re-streamed from L2 every timestep: the
// per-wave slice is 512 VGPRs, so register-caching is impossible and hoisting
// just spills to scratch. The opaque `zofs` (volatile asm, value always 0)
// makes the loads loop-variant so the compiler cannot hoist them.
// ---------------------------------------------------------------------------
__global__ void k_lstm_scan(const float* __restrict__ xgBlk,
                            const unsigned short* __restrict__ whh,
                            const int* __restrict__ mask,
                            float* __restrict__ pool_out,
                            float* __restrict__ final_out,
                            int T, int nSeqTiles, int nValid, int dirOff,
                            int reverse) {
  __shared__ unsigned short h_bf[16 * 256];   // 8 KB: h_{t-1} in bf16

  const int lane    = threadIdx.x & 31;
  const int wave    = threadIdx.x >> 5;
  const int sb      = blockIdx.x;
  const int mrow_hi = (lane >> 4) << 3;       // 0 or 8
  const int jcol    = wave * 32 + (lane & 15);

  for (int i = threadIdx.x; i < 16 * 256; i += blockDim.x) h_bf[i] = 0;
  __syncthreads();

  // Loop-invariant base pointers; all tile offsets below are immediates.
  const unsigned short* hptr  = h_bf + (lane & 15) * 256 + ((lane >> 4) << 3);
  const unsigned short* wbase = whh + (long)((wave * 2) * 16 + (lane & 15)) * 256
                                    + ((lane >> 4) << 3);
  const float* xbase = xgBlk + (((long)sb * 64 + wave * 2) * 32 + lane) * 8;
  const long tStride = (long)nSeqTiles * 64 * 32 * 8;   // floats per timestep

  v8f  cst[2]    = {v8f{}, v8f{}};
  v8f  pooled[2] = {v8f{}, v8f{}};
  float cnt[8];
#pragma unroll
  for (int r = 0; r < 8; ++r) cnt[r] = 0.f;

  for (int step = 0; step < T; ++step) {
    const int t = reverse ? (T - 1 - step) : step;
    const float* xstep = xbase + (long)t * tStride;

    // Opaque zero: re-materialized every iteration so W-fragment loads are
    // not loop-invariant (prevents hoist-and-spill of 512 VGPRs of weights).
    int zofs = 0;
    asm volatile("" : "+v"(zofs));
    const unsigned short* wstep = wbase + zofs;

    // ---- 1. accumulators <- blocked xg (one v8f per tile per lane)
    v8f acc[4][2];
#pragma unroll
    for (int g4 = 0; g4 < 4; ++g4) {
#pragma unroll
      for (int jj = 0; jj < 2; ++jj) {
        acc[g4][jj] = *(const v8f*)(xstep + g4 * 4096 + jj * 256);
      }
    }
    if (step + 1 < T) {   // prefetch next timestep's fragments
      const float* xnext = xstep + (reverse ? -tStride : tStride);
#pragma unroll
      for (int g4 = 0; g4 < 4; ++g4) {
#pragma unroll
        for (int jj = 0; jj < 2; ++jj)
          __builtin_prefetch(xnext + g4 * 4096 + jj * 256, 0, 1);
      }
    }

    // ---- 2. acc += h_{t-1} @ W_hh^T  (WMMA over K=256 in 8 chunks)
#pragma unroll
    for (int kc = 0; kc < 8; ++kc) {
      v16u afrag = load_frag_pair(hptr + kc * 32);
#pragma unroll
      for (int g4 = 0; g4 < 4; ++g4) {
#pragma unroll
        for (int jj = 0; jj < 2; ++jj) {
          v16u bfrag = load_frag_pair(wstep + g4 * 65536 + jj * 4096 + kc * 32);
          acc[g4][jj] = wmma_bf16_f32(afrag, bfrag, acc[g4][jj]);
        }
      }
    }
    __syncthreads();   // all waves done reading h_{t-1} from LDS

    // ---- 3. mask for this timestep (word level only)
    float mv[8];
#pragma unroll
    for (int r = 0; r < 8; ++r) mv[r] = 0.f;
    if (mask) {
#pragma unroll
      for (int r = 0; r < 8; ++r) {
        int m = r + mrow_hi;
        if (m < nValid) {
          mv[r] = (float)mask[(long)(sb * 16 + m) * T + t];
          cnt[r] += mv[r];
        }
      }
    }

    // ---- 4. gates + state update, write h_t to LDS (bf16)
#pragma unroll
    for (int jj = 0; jj < 2; ++jj) {
      int j = jcol + jj * 16;
#pragma unroll
      for (int r = 0; r < 8; ++r) {
        int m = r + mrow_hi;
        float iv = sigmoidf_fast(acc[0][jj][r]);
        float fv = sigmoidf_fast(acc[1][jj][r]);
        float gv = tanhf(acc[2][jj][r]);
        float ov = sigmoidf_fast(acc[3][jj][r]);
        float cv = fv * cst[jj][r] + iv * gv;
        cst[jj][r] = cv;
        float hv = ov * tanhf(cv);
        h_bf[m * 256 + j] = f32_to_bf16_rne(hv);
        pooled[jj][r] += mv[r] * hv;
        if (final_out && (t == T - 1) && (m < nValid))
          final_out[(long)(sb * 16 + m) * 512 + dirOff + j] = hv;
      }
    }
    __syncthreads();   // h_t visible to all waves for next step
  }

  // ---- masked mean pool (word level)
  if (pool_out) {
#pragma unroll
    for (int jj = 0; jj < 2; ++jj) {
      int j = jcol + jj * 16;
#pragma unroll
      for (int r = 0; r < 8; ++r) {
        int m = r + mrow_hi;
        if (m < nValid) {
          float d = cnt[r] > 0.f ? cnt[r] : 1.f;
          pool_out[(long)(sb * 16 + m) * 512 + dirOff + j] = pooled[jj][r] / d;
        }
      }
    }
  }
}

// ---------------------------------------------------------------------------
// Kernel: logits[8][5] = final[8][512] @ cls_w[5][512]^T + cls_b
// ---------------------------------------------------------------------------
__global__ void k_classifier(const float* __restrict__ fin,
                             const float* __restrict__ cw,
                             const float* __restrict__ cb,
                             float* __restrict__ out) {
  int i = threadIdx.x;
  if (i >= 40) return;
  int b = i / 5, c = i % 5;
  float s = cb[c];
  for (int j = 0; j < 512; ++j) s += fin[b * 512 + j] * cw[c * 512 + j];
  out[i] = s;
}

// ---------------------------------------------------------------------------
// Host launch
// ---------------------------------------------------------------------------
extern "C" void kernel_launch(void* const* d_in, const int* in_sizes, int n_in,
                              void* d_out, int out_size, void* d_ws, size_t ws_size,
                              hipStream_t stream) {
  (void)in_sizes; (void)n_in; (void)out_size; (void)ws_size;

  const float* hs     = (const float*)d_in[0];    // [8,32,128,768]
  const int*   amask  = (const int*)  d_in[1];    // [8,32,128]
  const float* w_ih_f = (const float*)d_in[2];    // [1024,768]
  const float* w_hh_f = (const float*)d_in[3];    // [1024,256]
  const float* b_f    = (const float*)d_in[4];    // [1024]
  const float* w_ih_b = (const float*)d_in[5];
  const float* w_hh_b = (const float*)d_in[6];
  const float* b_b    = (const float*)d_in[7];
  const float* s_ih_f = (const float*)d_in[8];    // [1024,512]
  const float* s_hh_f = (const float*)d_in[9];    // [1024,256]
  const float* sb_f   = (const float*)d_in[10];
  const float* s_ih_b = (const float*)d_in[11];
  const float* s_hh_b = (const float*)d_in[12];
  const float* sb_b   = (const float*)d_in[13];
  const float* cls_w  = (const float*)d_in[14];   // [5,512]
  const float* cls_b  = (const float*)d_in[15];   // [5]
  float* out = (float*)d_out;                     // [8,5]

  // Workspace carve-up (256B aligned)
  char* wp = (char*)d_ws;
  auto carve = [&](size_t bytes) -> char* {
    char* p = wp;
    wp += (bytes + 255) & ~(size_t)255;
    return p;
  };
  unsigned short* x_bf    = (unsigned short*)carve((size_t)32768 * 768 * 2);
  unsigned short* wihf_bf = (unsigned short*)carve(1024 * 768 * 2);
  unsigned short* wihb_bf = (unsigned short*)carve(1024 * 768 * 2);
  unsigned short* whhf_bf = (unsigned short*)carve(1024 * 256 * 2);
  unsigned short* whhb_bf = (unsigned short*)carve(1024 * 256 * 2);
  unsigned short* sihf_bf = (unsigned short*)carve(1024 * 512 * 2);
  unsigned short* sihb_bf = (unsigned short*)carve(1024 * 512 * 2);
  unsigned short* shhf_bf = (unsigned short*)carve(1024 * 256 * 2);
  unsigned short* shhb_bf = (unsigned short*)carve(1024 * 256 * 2);
  float* xg_f   = (float*)carve((size_t)32768 * 1024 * 4);   // blocked
  float* xg_b   = (float*)carve((size_t)32768 * 1024 * 4);   // blocked
  float* sentem = (float*)carve((size_t)256 * 512 * 4);
  unsigned short* sentem_bf = (unsigned short*)carve((size_t)256 * 512 * 2);
  float* sxg_f  = (float*)carve((size_t)512 * 1024 * 4);     // blocked, padded
  float* sxg_b  = (float*)carve((size_t)512 * 1024 * 4);     // blocked, padded
  float* final8 = (float*)carve((size_t)8 * 512 * 4);

  auto conv = [&](const float* src, unsigned short* dst, int n) {
    int n8 = n / 8;
    k_f32_to_bf16x8<<<dim3((n8 + 255) / 256), dim3(256), 0, stream>>>(src, dst, n8);
  };
  conv(hs,     x_bf,    32768 * 768);
  conv(w_ih_f, wihf_bf, 1024 * 768);
  conv(w_ih_b, wihb_bf, 1024 * 768);
  conv(w_hh_f, whhf_bf, 1024 * 256);
  conv(w_hh_b, whhb_bf, 1024 * 256);
  conv(s_ih_f, sihf_bf, 1024 * 512);
  conv(s_ih_b, sihb_bf, 1024 * 512);
  conv(s_hh_f, shhf_bf, 1024 * 256);
  conv(s_hh_b, shhb_bf, 1024 * 256);

  // Word-level input projections (t-major blocked output)
  {
    dim3 grid(32768 / 16);
    k_gemm_bf16_blk<768><<<grid, dim3(256), 0, stream>>>(x_bf, wihf_bf, b_f,
                                                         xg_f, 256, 128, 256);
    k_gemm_bf16_blk<768><<<grid, dim3(256), 0, stream>>>(x_bf, wihb_bf, b_b,
                                                         xg_b, 256, 128, 256);
  }

  // Word-level recurrences (256 sequences, T=128) + masked mean pooling
  {
    dim3 grid(16);
    k_lstm_scan<<<grid, dim3(256), 0, stream>>>(xg_f, whhf_bf, amask, sentem,
                                                nullptr, 128, 16, 16, 0, 0);
    k_lstm_scan<<<grid, dim3(256), 0, stream>>>(xg_b, whhb_bf, amask, sentem,
                                                nullptr, 128, 16, 16, 256, 1);
  }

  // Sentence-level: convert pooled embeddings, project (padded to 16 seqs)
  conv(sentem, sentem_bf, 256 * 512);
  {
    dim3 grid(512 / 16);
    k_gemm_bf16_blk<512><<<grid, dim3(256), 0, stream>>>(sentem_bf, sihf_bf,
                                                         sb_f, sxg_f, 16, 32, 8);
    k_gemm_bf16_blk<512><<<grid, dim3(256), 0, stream>>>(sentem_bf, sihb_bf,
                                                         sb_b, sxg_b, 16, 32, 8);
  }

  // Sentence-level recurrences (8 valid seqs of 16, T=32); capture h at t==T-1
  {
    dim3 grid(1);
    k_lstm_scan<<<grid, dim3(256), 0, stream>>>(sxg_f, shhf_bf, nullptr, nullptr,
                                                final8, 32, 1, 8, 0, 0);
    k_lstm_scan<<<grid, dim3(256), 0, stream>>>(sxg_b, shhb_bf, nullptr, nullptr,
                                                final8, 32, 1, 8, 256, 1);
  }

  // Classifier
  k_classifier<<<dim3(1), dim3(64), 0, stream>>>(final8, cls_w, cls_b, out);
}
